// CodebookLayer_13005160973060
// MI455X (gfx1250) — matrix operand order repaired and checked by reading, not compiled, not run on hardware
//
#include <hip/hip_runtime.h>
#include <stdint.h>

// ---------- CDNA5 vector types ----------
typedef __attribute__((ext_vector_type(16))) __bf16   v16bf;
typedef __attribute__((ext_vector_type(8)))  float    v8f;
typedef __attribute__((ext_vector_type(4)))  uint32_t v4u;
typedef __attribute__((ext_vector_type(4)))  int      v4i;

#define D_DIM   1024      // feature dim (K)
#define C_DIM   16384     // codebook entries (N)
#define ROWS    16384     // 4*4096 flattened x rows (M)
#define M_TILE  64        // rows per block
#define N_PANEL 128       // codebook rows per sweep step
#define K_STEP  32        // bf16 WMMA K
#define KSTEPS  (D_DIM / K_STEP)

// double-buffered LDS arena layout (bytes)
#define OFF_AH  0         // A hi : 64 rows * 32 elts * 2B = 4096
#define OFF_AL  4096      // A lo : 4096
#define OFF_BH  8192      // B hi : 128 rows * 32 elts * 2B = 8192
#define OFF_BL  16384     // B lo : 8192
#define BUF_BYTES 24576   // per buffer
// reduction scratch aliases buffer 0 (only used after final K-step barrier)
#define OFF_RMIN 0        // 64*32*4 = 8192
#define OFF_RIDX 8192     // 8192
#define OFF_IDX  16384    // 256

#define AS1 __attribute__((address_space(1)))
#define AS3 __attribute__((address_space(3)))

// fp32 -> bf16 round-to-nearest-even (bit twiddle; avoids __bf16 arithmetic)
static __device__ __forceinline__ uint16_t f2bf_rne(float f) {
  uint32_t u = __float_as_uint(f);
  uint32_t r = u + 0x7FFFu + ((u >> 16) & 1u);
  return (uint16_t)(r >> 16);
}
static __device__ __forceinline__ float bf2f(uint16_t b) {
  return __uint_as_float(((uint32_t)b) << 16);
}

union Frag16 { v4u u[2]; v16bf bf; };

// async global->LDS 16B copy (GLOBAL_LOAD_ASYNC_TO_LDS_B128, ASYNCcnt-tracked)
static __device__ __forceinline__ void async_cp16(const uint16_t* g, char* l) {
  __builtin_amdgcn_global_load_async_to_lds_b128(
      (AS1 v4i*)(uintptr_t)g, (AS3 v4i*)l, 0, 0);
}
static __device__ __forceinline__ void wait_async_all() {
  asm volatile("s_wait_asynccnt 0x0" ::: "memory");
}

// ---------- prep: fp32 -> (bf16 hi, bf16 lo) split ----------
__global__ __launch_bounds__(256) void prep_split(const float* __restrict__ src,
                                                  uint16_t* __restrict__ hi,
                                                  uint16_t* __restrict__ lo,
                                                  int n) {
  int i = blockIdx.x * blockDim.x + threadIdx.x;
  int stride = gridDim.x * blockDim.x;
  for (; i < n; i += stride) {
    float v = src[i];
    uint16_t h = f2bf_rne(v);
    hi[i] = h;
    lo[i] = f2bf_rne(v - bf2f(h));
  }
}

// ---------- prep: codebook row squared norms ----------
__global__ __launch_bounds__(256) void row_norms(const float* __restrict__ cb,
                                                 float* __restrict__ c2) {
  __shared__ float red[256];
  const int row = blockIdx.x;
  const float* p = cb + (size_t)row * D_DIM;
  float s = 0.f;
  for (int j = threadIdx.x; j < D_DIM; j += 256) { float v = p[j]; s += v * v; }
  red[threadIdx.x] = s;
  __syncthreads();
  for (int o = 128; o > 0; o >>= 1) {
    if (threadIdx.x < o) red[threadIdx.x] += red[threadIdx.x + o];
    __syncthreads();
  }
  if (threadIdx.x == 0) c2[row] = red[0];
}

// ---------- main: WMMA distance sweep + argmin + gather ----------
__global__ __launch_bounds__(256) void snap_wmma(
    const uint16_t* __restrict__ xh, const uint16_t* __restrict__ xl,
    const uint16_t* __restrict__ ch, const uint16_t* __restrict__ cl,
    const float* __restrict__ c2, const float* __restrict__ codebook,
    float* __restrict__ out) {
  __shared__ __align__(16) char smem[2 * BUF_BYTES];  // 48 KB arena

  const int tid  = threadIdx.x;
  const int wave = tid >> 5;
  const int lane = tid & 31;
  const int mblock = blockIdx.x * M_TILE;

  const int mStrip = (wave & 3) * 16;  // 16-row M strip owned by this wave
  const int nHalf  = wave >> 2;        // which half of the 128-wide N panel

  // cooperative async staging coordinates
  const int aRow = tid >> 2, aOff = (tid & 3) * 8;   // 64 rows x 4 thr x 16B
  const int bRow = tid >> 1, bOff = (tid & 1) * 16;  // 128 rows x 2 thr x 32B

  // per-lane WMMA fragment addressing (ISA 7.12.2 layouts, wave32)
  const int al15   = lane & 15;
  const int aChunk = (lane < 16) ? 0 : 8;    // A: lane pair covers K chunks {c, c+16}
  const int bChunk = (lane < 16) ? 0 : 8;    // B: lane pair covers K chunks {c, c+16}

  float minv[8];
  int   mini[8];
#pragma unroll
  for (int r = 0; r < 8; ++r) { minv[r] = 3.4e38f; mini[r] = 0; }

  const v8f vzero = {0.f, 0.f, 0.f, 0.f, 0.f, 0.f, 0.f, 0.f};

  for (int nOut = 0; nOut < C_DIM / N_PANEL; ++nOut) {
    const int nBase = nOut * N_PANEL;
    v8f acc[4];
#pragma unroll
    for (int t = 0; t < 4; ++t) acc[t] = vzero;

    // prologue: async-stage K panel 0 into buffer 0
    {
      const size_t ga = (size_t)(mblock + aRow) * D_DIM + aOff;
      const size_t gb = (size_t)(nBase + bRow) * D_DIM + bOff;
      char* b0 = smem;
      async_cp16(xh + ga, b0 + OFF_AH + aRow * 64 + aOff * 2);
      async_cp16(xl + ga, b0 + OFF_AL + aRow * 64 + aOff * 2);
      async_cp16(ch + gb,     b0 + OFF_BH + bRow * 64 + bOff * 2);
      async_cp16(ch + gb + 8, b0 + OFF_BH + bRow * 64 + bOff * 2 + 16);
      async_cp16(cl + gb,     b0 + OFF_BL + bRow * 64 + bOff * 2);
      async_cp16(cl + gb + 8, b0 + OFF_BL + bRow * 64 + bOff * 2 + 16);
    }
    wait_async_all();
    __syncthreads();

    for (int ks = 0; ks < KSTEPS; ++ks) {
      char* cur = smem + (ks & 1) * BUF_BYTES;
      // kick off async staging of the NEXT K panel into the other buffer;
      // the async engine overlaps this with the WMMA work below
      if (ks + 1 < KSTEPS) {
        char* nxt = smem + ((ks + 1) & 1) * BUF_BYTES;
        const int kNext = (ks + 1) * K_STEP;
        const size_t ga = (size_t)(mblock + aRow) * D_DIM + kNext + aOff;
        const size_t gb = (size_t)(nBase + bRow) * D_DIM + kNext + bOff;
        async_cp16(xh + ga, nxt + OFF_AH + aRow * 64 + aOff * 2);
        async_cp16(xl + ga, nxt + OFF_AL + aRow * 64 + aOff * 2);
        async_cp16(ch + gb,     nxt + OFF_BH + bRow * 64 + bOff * 2);
        async_cp16(ch + gb + 8, nxt + OFF_BH + bRow * 64 + bOff * 2 + 16);
        async_cp16(cl + gb,     nxt + OFF_BL + bRow * 64 + bOff * 2);
        async_cp16(cl + gb + 8, nxt + OFF_BL + bRow * 64 + bOff * 2 + 16);
      }

      // A fragments (this wave's 16 M rows), hi and lo planes
      Frag16 ah, al;
      const uint16_t* rowAh =
          (const uint16_t*)(cur + OFF_AH) + (mStrip + al15) * K_STEP;
      const uint16_t* rowAl =
          (const uint16_t*)(cur + OFF_AL) + (mStrip + al15) * K_STEP;
      ah.u[0] = *(const v4u*)(rowAh + aChunk);
      ah.u[1] = *(const v4u*)(rowAh + aChunk + 16);
      al.u[0] = *(const v4u*)(rowAl + aChunk);
      al.u[1] = *(const v4u*)(rowAl + aChunk + 16);

      // B fragments for all 4 tiles (hi+lo) loaded up front so the WMMA
      // passes below see 4 independent accumulators back-to-back (hides
      // the 5-slot WMMA->WMMA RAW hazard on the XDL pipe)
      Frag16 bh[4], bl[4];
#pragma unroll
      for (int t = 0; t < 4; ++t) {
        const int nRow = (nHalf * 4 + t) * 16 + al15;
        const uint16_t* rowBh =
            (const uint16_t*)(cur + OFF_BH) + nRow * K_STEP + bChunk;
        const uint16_t* rowBl =
            (const uint16_t*)(cur + OFF_BL) + nRow * K_STEP + bChunk;
        bh[t].u[0] = *(const v4u*)(rowBh);
        bh[t].u[1] = *(const v4u*)(rowBh + 16);
        bl[t].u[0] = *(const v4u*)(rowBl);
        bl[t].u[1] = *(const v4u*)(rowBl + 16);
      }

      // split-precision dot: hi*hi + lo*hi + hi*lo, fp32 accumulate.
      // Three passes over t => same-acc WMMA reuse distance of 4.
#pragma unroll
      for (int t = 0; t < 4; ++t)
        acc[t] = __builtin_amdgcn_wmma_f32_16x16x32_bf16(
            false, ah.bf, false, bh[t].bf, (short)0, acc[t], false, false);
#pragma unroll
      for (int t = 0; t < 4; ++t)
        acc[t] = __builtin_amdgcn_wmma_f32_16x16x32_bf16(
            false, al.bf, false, bh[t].bf, (short)0, acc[t], false, false);
#pragma unroll
      for (int t = 0; t < 4; ++t)
        acc[t] = __builtin_amdgcn_wmma_f32_16x16x32_bf16(
            false, ah.bf, false, bl[t].bf, (short)0, acc[t], false, false);

      // ensure our async copies into the other buffer completed, then
      // barrier so every wave's panel data is visible for the next step
      wait_async_all();
      __syncthreads();
    }

    // score = c2[n] - 2*dot; fold into running per-row argmin
#pragma unroll
    for (int t = 0; t < 4; ++t) {
      const int n = nBase + (nHalf * 4 + t) * 16 + al15;
      const float cc = c2[n];
#pragma unroll
      for (int r = 0; r < 8; ++r) {
        const float s = cc - 2.0f * acc[t][r];
        if (s < minv[r]) { minv[r] = s; mini[r] = n; }
        else if (s == minv[r] && n < mini[r]) { mini[r] = n; }
      }
    }
  }

  // cross-lane/cross-wave argmin merge through LDS (ties -> smallest index)
  // (aliases staging buffer 0 -- safe: last LDS reads were barrier-ordered)
  float* sRedMin = (float*)(smem + OFF_RMIN);
  int*   sRedIdx = (int*)(smem + OFF_RIDX);
  int*   sIdx    = (int*)(smem + OFF_IDX);
#pragma unroll
  for (int r = 0; r < 8; ++r) {
    const int ml = mStrip + r + ((lane >> 4) << 3);  // local M row 0..63
    const int bucket = nHalf * 16 + al15;            // 32 buckets per row
    sRedMin[ml * 32 + bucket] = minv[r];
    sRedIdx[ml * 32 + bucket] = mini[r];
  }
  __syncthreads();
  if (tid < M_TILE) {
    float bm = sRedMin[tid * 32];
    int   bi = sRedIdx[tid * 32];
    for (int j = 1; j < 32; ++j) {
      const float v = sRedMin[tid * 32 + j];
      const int  ii = sRedIdx[tid * 32 + j];
      if (v < bm || (v == bm && ii < bi)) { bm = v; bi = ii; }
    }
    sIdx[tid] = bi;
  }
  __syncthreads();

  // gather winning codebook rows -> output (fp32, 4KB per row)
  for (int row = wave; row < M_TILE; row += 8) {
    const v4u* src = (const v4u*)(codebook + (size_t)sIdx[row] * D_DIM);
    v4u* dst = (v4u*)(out + (size_t)(mblock + row) * D_DIM);
    for (int j = lane; j < D_DIM / 4; j += 32) dst[j] = src[j];
  }
}

// ---------- host launcher ----------
extern "C" void kernel_launch(void* const* d_in, const int* in_sizes, int n_in,
                              void* d_out, int out_size, void* d_ws, size_t ws_size,
                              hipStream_t stream) {
  (void)in_sizes; (void)n_in; (void)out_size; (void)ws_size;
  const float* x  = (const float*)d_in[0];  // [4,4096,1024] fp32
  const float* cb = (const float*)d_in[1];  // [16384,1024] fp32
  float* out = (float*)d_out;               // [4,4096,1024] fp32

  const size_t nCb = (size_t)C_DIM * D_DIM;  // 16,777,216
  const size_t nX  = (size_t)ROWS * D_DIM;   // 16,777,216
  uint16_t* cb_hi = (uint16_t*)d_ws;         // 32 MB
  uint16_t* cb_lo = cb_hi + nCb;             // 32 MB
  uint16_t* x_hi  = cb_lo + nCb;             // 32 MB
  uint16_t* x_lo  = x_hi + nX;               // 32 MB
  float*    c2    = (float*)(x_lo + nX);     // 64 KB

  prep_split<<<2048, 256, 0, stream>>>(cb, cb_hi, cb_lo, (int)nCb);
  prep_split<<<2048, 256, 0, stream>>>(x, x_hi, x_lo, (int)nX);
  row_norms<<<C_DIM, 256, 0, stream>>>(cb, c2);
  snap_wmma<<<ROWS / M_TILE, 256, 0, stream>>>(x_hi, x_lo, cb_hi, cb_lo, c2, cb, out);
}
